// AdditiveAttention_72189810311179
// MI455X (gfx1250) — matrix operand order repaired and checked by reading, not compiled, not run on hardware
//
#include <hip/hip_runtime.h>

#define B_ 32
#define S_ 4096
#define H_ 512
#define CHUNK 256
#define NCHUNK (S_ / CHUNK)     // 16
#define MPASS 128
#define NPASS (CHUNK / MPASS)   // 2
#define KSLAB 32
#define NKSLAB (H_ / KSLAB)     // 16
#define APAD 40                 // halves per row in LDS tiles (80B, 16B-aligned)
#define LOG2E 1.44269504088896f

typedef _Float16 v16h __attribute__((ext_vector_type(16)));
typedef _Float16 v8h  __attribute__((ext_vector_type(8)));
typedef _Float16 h2   __attribute__((ext_vector_type(2)));
typedef float    v8f  __attribute__((ext_vector_type(8)));
typedef float    v4f  __attribute__((ext_vector_type(4)));

__device__ __forceinline__ float fast_tanh(float x) {
    // tanh(x) = 1 - 2/(exp(2x)+1); v_exp_f32 is exp2, v_rcp_f32 approx rcp.
    float e = __builtin_amdgcn_exp2f(x * (2.0f * LOG2E));
    return 1.0f - 2.0f * __builtin_amdgcn_rcpf(e + 1.0f);
}

// ---------------------------------------------------------------------------
// Kernel 0: qv[b,n] = query[b,:] @ Wq[:,n] + bq[n] + bv[n]
// ---------------------------------------------------------------------------
__global__ __launch_bounds__(256) void qproj_kernel(
    const float* __restrict__ query, const float* __restrict__ Wq,
    const float* __restrict__ bq, const float* __restrict__ bv,
    float* __restrict__ qv)
{
    int gid = blockIdx.x * 256 + threadIdx.x;
    int b = gid >> 9;
    int n = gid & (H_ - 1);
    const float* q = query + b * H_;
    float s = bq[n] + bv[n];
#pragma unroll 4
    for (int h = 0; h < H_; ++h) s += q[h] * Wq[h * H_ + n];
    qv[gid] = s;
}

// ---------------------------------------------------------------------------
// Kernel 1: per (b, 256-row chunk): WMMA f16 GEMM values@Wv (f32 acc),
// tanh epilogue, score reduction, chunk softmax stats + partial context.
// ---------------------------------------------------------------------------
__global__ __launch_bounds__(256, 1) void attn_scores_kernel(
    const float* __restrict__ values, const int* __restrict__ mask,
    const float* __restrict__ Wv, const float* __restrict__ wc,
    const float* __restrict__ bc, const float* __restrict__ qv,
    float* __restrict__ scores_g, float* __restrict__ ml_g,
    float* __restrict__ cpart_g)
{
    __shared__ _Float16 Ah[MPASS * APAD];   // values tile [row][k], f16
    __shared__ _Float16 Bt[H_ * APAD];      // Wv tile transposed [n][k], f16
    __shared__ float score_c[CHUNK];
    __shared__ float wred[8];
    __shared__ float sStat[2];

    const int t = threadIdx.x;
    const int b = blockIdx.x / NCHUNK;
    const int c = blockIdx.x % NCHUNK;
    const int chunk_base = c * CHUNK;
    const int lane = t & 31;
    const int wv = t >> 5;          // wave 0..7, owns columns [wv*64, wv*64+64)
    const int hi = lane >> 4;
    const int lx = lane & 15;

    score_c[t] = 0.0f;

    // per-lane column constants: q-projection+bv, and wc
    float qn[4], wn[4];
#pragma unroll
    for (int nt = 0; nt < 4; ++nt) {
        int ncol = wv * 64 + nt * 16 + lx;
        qn[nt] = qv[b * H_ + ncol];
        wn[nt] = wc[ncol];
    }
    __syncthreads();

    for (int p = 0; p < NPASS; ++p) {
        const int row0 = chunk_base + p * MPASS;
        const float* vrow = values + ((size_t)b * S_ + row0) * H_;

        v8f acc[8][4];
#pragma unroll
        for (int i = 0; i < 8; ++i)
#pragma unroll
            for (int j = 0; j < 4; ++j)
#pragma unroll
                for (int r = 0; r < 8; ++r) acc[i][j][r] = 0.0f;

        for (int ks = 0; ks < NKSLAB; ++ks) {
            const int k0 = ks * KSLAB;
            __syncthreads();
            // ---- stage A: 128 rows x 32 k of values, f32 -> f16 ----
            {
                int row = t >> 1, koff = (t & 1) * 16;
                const float* src = vrow + (size_t)row * H_ + k0 + koff;
                _Float16* dst = &Ah[row * APAD + koff];
#pragma unroll
                for (int i = 0; i < 16; ++i) dst[i] = (_Float16)src[i];
            }
            // ---- stage B transposed: Wv[k0+k][n] -> Bt[n][k] (pack k-pairs) ----
            {
                const v4f* Wv4 = (const v4f*)(Wv + (size_t)k0 * H_);
#pragma unroll
                for (int i = 0; i < 8; ++i) {
                    int idx = t + 256 * i;       // 0..2047 -> (k2, n4)
                    int k2 = idx >> 7;           // 0..15 (pair of k rows)
                    int n4 = idx & 127;          // group of 4 columns
                    v4f f0 = Wv4[(size_t)(2 * k2) * (H_ / 4) + n4];
                    v4f f1 = Wv4[(size_t)(2 * k2 + 1) * (H_ / 4) + n4];
#pragma unroll
                    for (int cc = 0; cc < 4; ++cc) {
                        h2 pk; pk[0] = (_Float16)f0[cc]; pk[1] = (_Float16)f1[cc];
                        *(h2*)&Bt[(n4 * 4 + cc) * APAD + 2 * k2] = pk;
                    }
                }
            }
            __syncthreads();

            // ---- B fragments for this wave's 4 n-tiles ----
            // B layout: lane half hi: K runs [hi*8, hi*8+8) and [16+hi*8, 16+hi*8+8).
            v16h bf[4];
#pragma unroll
            for (int nt = 0; nt < 4; ++nt) {
                int ncol = wv * 64 + nt * 16 + lx;
                const _Float16* pb = &Bt[ncol * APAD + hi * 8];
                v8h blo = *(const v8h*)pb, bhi = *(const v8h*)(pb + 16);
#pragma unroll
                for (int j = 0; j < 8; ++j) { bf[nt][j] = blo[j]; bf[nt][8 + j] = bhi[j]; }
            }
            // ---- A fragments + WMMA ----
            // A layout: lane half hi: K runs [hi*8, hi*8+8) and [16+hi*8, 16+hi*8+8).
#pragma unroll
            for (int mt = 0; mt < 8; ++mt) {
                int mrow = mt * 16 + lx;
                v8h alo = *(const v8h*)&Ah[mrow * APAD + hi * 8];
                v8h ahi2 = *(const v8h*)&Ah[mrow * APAD + 16 + hi * 8];
                v16h af;
#pragma unroll
                for (int j = 0; j < 8; ++j) { af[j] = alo[j]; af[8 + j] = ahi2[j]; }
#pragma unroll
                for (int nt = 0; nt < 4; ++nt) {
                    acc[mt][nt] = __builtin_amdgcn_wmma_f32_16x16x32_f16(
                        false, af, false, bf[nt], (short)0, acc[mt][nt], false, false);
                }
            }
        }

        // ---- epilogue: tanh, * wc, reduce over n -> chunk scores ----
#pragma unroll
        for (int mt = 0; mt < 8; ++mt) {
            float rp[8];
#pragma unroll
            for (int r = 0; r < 8; ++r) rp[r] = 0.0f;
#pragma unroll
            for (int nt = 0; nt < 4; ++nt) {
#pragma unroll
                for (int r = 0; r < 8; ++r) {
                    float th = fast_tanh(acc[mt][nt][r] + qn[nt]);
                    rp[r] += wn[nt] * th;
                }
            }
#pragma unroll
            for (int m = 1; m < 16; m <<= 1)
#pragma unroll
                for (int r = 0; r < 8; ++r) rp[r] += __shfl_xor(rp[r], m, 32);
            if (lx == 0) {
                int mbase = p * MPASS + mt * 16 + 8 * hi;
#pragma unroll
                for (int r = 0; r < 8; ++r) atomicAdd(&score_c[mbase + r], rp[r]);
            }
        }
    }
    __syncthreads();

    // ---- finalize chunk scores: +bc, mask, write raw, chunk max/sumexp ----
    const float bc0 = bc[0];
    float sc = score_c[t] + bc0;
    if (mask[b * S_ + chunk_base + t] == 0) sc = -__builtin_inff();
    scores_g[(size_t)b * S_ + chunk_base + t] = sc;

    float v = sc;
#pragma unroll
    for (int m = 16; m >= 1; m >>= 1) v = fmaxf(v, __shfl_xor(v, m, 32));
    if (lane == 0) wred[wv] = v;
    __syncthreads();
    if (t == 0) {
        float mx = wred[0];
        for (int i = 1; i < 8; ++i) mx = fmaxf(mx, wred[i]);
        sStat[0] = mx;
    }
    __syncthreads();
    float mC = sStat[0];
    float w = __builtin_amdgcn_exp2f((sc - mC) * LOG2E);
    score_c[t] = w;                      // overwrite with softmax weight
    float sv = w;
#pragma unroll
    for (int m = 16; m >= 1; m >>= 1) sv += __shfl_xor(sv, m, 32);
    __syncthreads();                     // wred reuse safe after this
    if (lane == 0) wred[wv] = sv;
    __syncthreads();
    if (t == 0) {
        float L = 0.f;
        for (int i = 0; i < 8; ++i) L += wred[i];
        sStat[1] = L;
    }
    __syncthreads();

    // ---- partial context: c[h] = sum_m w[m] * values[row m][h] (L2-hot) ----
    float c0 = 0.f, c1 = 0.f;
    const float* vb = values + ((size_t)b * S_ + chunk_base) * H_;
    for (int m2 = 0; m2 < CHUNK; ++m2) {
        float ww = score_c[m2];
        c0 += ww * vb[(size_t)m2 * H_ + t];
        c1 += ww * vb[(size_t)m2 * H_ + t + 256];
    }
    size_t ci = (size_t)(b * NCHUNK + c) * H_;
    cpart_g[ci + t] = c0;
    cpart_g[ci + t + 256] = c1;
    if (t == 0) {
        ml_g[2 * (b * NCHUNK + c) + 0] = mC;
        ml_g[2 * (b * NCHUNK + c) + 1] = sStat[1];
    }
}

// ---------------------------------------------------------------------------
// Kernel 2: per b: combine chunk partials -> context, write normalized attn.
// ---------------------------------------------------------------------------
__global__ __launch_bounds__(256) void attn_finalize_kernel(
    const float* __restrict__ scores_g, const float* __restrict__ ml_g,
    const float* __restrict__ cpart_g,
    float* __restrict__ out_ctx, float* __restrict__ out_attn)
{
    __shared__ float coef[NCHUNK];
    __shared__ float sML[2];
    int b = blockIdx.x, t = threadIdx.x;
    if (t == 0) {
        float M = -__builtin_inff();
        for (int i = 0; i < NCHUNK; ++i) M = fmaxf(M, ml_g[2 * (b * NCHUNK + i)]);
        float L = 0.f;
        for (int i = 0; i < NCHUNK; ++i) {
            float cf = __builtin_amdgcn_exp2f((ml_g[2 * (b * NCHUNK + i)] - M) * LOG2E);
            coef[i] = cf;
            L += cf * ml_g[2 * (b * NCHUNK + i) + 1];
        }
        sML[0] = M; sML[1] = L;
    }
    __syncthreads();
    float M = sML[0];
    float invL = 1.0f / sML[1];
#pragma unroll
    for (int hh = 0; hh < 2; ++hh) {
        int h = t + hh * 256;
        float cacc = 0.f;
        for (int i = 0; i < NCHUNK; ++i)
            cacc += coef[i] * cpart_g[(size_t)(b * NCHUNK + i) * H_ + h];
        out_ctx[(size_t)b * H_ + h] = cacc * invL;
    }
    for (int s = t; s < S_; s += 256) {
        float sc = scores_g[(size_t)b * S_ + s];
        out_attn[(size_t)b * S_ + s] = __builtin_amdgcn_exp2f((sc - M) * LOG2E) * invL;
    }
}

// ---------------------------------------------------------------------------
extern "C" void kernel_launch(void* const* d_in, const int* in_sizes, int n_in,
                              void* d_out, int out_size, void* d_ws, size_t ws_size,
                              hipStream_t stream)
{
    (void)in_sizes; (void)n_in; (void)out_size; (void)ws_size;
    const float* query  = (const float*)d_in[0];
    const float* values = (const float*)d_in[1];
    const int*   mask   = (const int*)d_in[2];
    const float* Wq     = (const float*)d_in[3];
    const float* bq     = (const float*)d_in[4];
    const float* Wv     = (const float*)d_in[5];
    const float* bv     = (const float*)d_in[6];
    const float* wc     = (const float*)d_in[7];
    const float* bc     = (const float*)d_in[8];

    float* out_ctx  = (float*)d_out;                  // [B,H]
    float* out_attn = out_ctx + (size_t)B_ * H_;      // [B,S]

    float* ws     = (float*)d_ws;
    float* qv     = ws;                               // B*H
    float* scores = qv + B_ * H_;                     // B*S
    float* ml     = scores + (size_t)B_ * S_;         // B*NCHUNK*2
    float* cpart  = ml + B_ * NCHUNK * 2;             // B*NCHUNK*H

    qproj_kernel<<<B_ * H_ / 256, 256, 0, stream>>>(query, Wq, bq, bv, qv);
    attn_scores_kernel<<<B_ * NCHUNK, 256, 0, stream>>>(
        values, mask, Wv, wc, bc, qv, scores, ml, cpart);
    attn_finalize_kernel<<<B_, 256, 0, stream>>>(scores, ml, cpart, out_ctx, out_attn);
}
